// VectorQuantizer_274877906975
// MI455X (gfx1250) — matrix operand order repaired and checked by reading, not compiled, not run on hardware
//
#include <hip/hip_runtime.h>
#include <math.h>

// ---------------- types ----------------
typedef __attribute__((ext_vector_type(16))) _Float16 v16h;
typedef __attribute__((ext_vector_type(8)))  _Float16 v8h;
typedef __attribute__((ext_vector_type(8)))  float    v8f;

union F16x16 { v16h v; v8h h[2]; };

#define KCODES 8192
#define DDIM   256
#define LLEN   4096
#define NTOK   16384   // B*L = 4*4096

// ---------------------------------------------------------------
// K1: normalize codebook rows.
//   cn  [K,D] f32   (needed for q0 gather in rotation)
//   cnT [D,K] f16   (B-operand for WMMA, k-contiguous rows)
// Block = 256 threads handles 32 codes.
// ---------------------------------------------------------------
__global__ void k_cbnorm(const float* __restrict__ cb,
                         float* __restrict__ cn,
                         _Float16* __restrict__ cnT,
                         float* __restrict__ loss_slot) {
    __shared__ float part[32][8];
    __shared__ float inv[32];
    const int k0  = blockIdx.x * 32;
    const int tid = threadIdx.x;
    const int kk  = tid >> 3;      // 0..31 code within block
    const int dp  = tid & 7;       // 0..7  d-phase
    const float* row = cb + (size_t)(k0 + kk) * DDIM;
    float s = 0.f;
    for (int d = dp; d < DDIM; d += 8) { float v = row[d]; s += v * v; }
    part[kk][dp] = s;
    __syncthreads();
    if (tid < 32) {
        float t = 0.f;
        #pragma unroll
        for (int j = 0; j < 8; ++j) t += part[tid][j];
        inv[tid] = 1.0f / fmaxf(sqrtf(t), 1e-12f);
    }
    __syncthreads();
    // f32 normalized codebook, coalesced-ish writes
    {
        float iv = inv[kk];
        float* dst = cn + (size_t)(k0 + kk) * DDIM;
        for (int d = dp; d < DDIM; d += 8) dst[d] = row[d] * iv;
    }
    // f16 transposed codebook: lanes sweep k (coalesced stores)
    {
        const int kx = tid & 31;   // k within block, lane-fast
        const int dw = tid >> 5;   // 0..7
        float iv = inv[kx];
        const float* rx = cb + (size_t)(k0 + kx) * DDIM;
        for (int d = dw; d < DDIM; d += 8)
            cnT[(size_t)d * KCODES + k0 + kx] = (_Float16)(rx[d] * iv);
    }
    if (blockIdx.x == 0 && tid == 0) *loss_slot = 0.0f;
}

// ---------------------------------------------------------------
// K2: build xn f16 [N,D] row-major (token-major) from x [B,D,L].
// Block = 256 threads handles 32 tokens (consecutive l, one b).
// Coalesced loads along l, LDS transpose, coalesced stores along d.
// ---------------------------------------------------------------
__global__ void k_xnorm(const float* __restrict__ x,
                        _Float16* __restrict__ xnh) {
    __shared__ float tile[DDIM * 33];   // [d][token], padded
    __shared__ float part[8][32];
    __shared__ float inv[32];
    const int n0   = blockIdx.x * 32;
    const int b    = n0 / LLEN;
    const int l0   = n0 % LLEN;
    const int tid  = threadIdx.x;
    const int lane = tid & 31;          // token within tile
    const int w    = tid >> 5;          // 0..7
    const float* xb = x + (size_t)b * DDIM * LLEN + l0;
    float s = 0.f;
    #pragma unroll 4
    for (int i = 0; i < 32; ++i) {
        int d = w + 8 * i;
        float v = xb[(size_t)d * LLEN + lane];
        tile[d * 33 + lane] = v;
        s += v * v;
    }
    part[w][lane] = s;
    __syncthreads();
    if (tid < 32) {
        float t = 0.f;
        #pragma unroll
        for (int j = 0; j < 8; ++j) t += part[j][tid];
        inv[tid] = 1.0f / fmaxf(sqrtf(t), 1e-12f);
    }
    __syncthreads();
    for (int mi = w; mi < 32; mi += 8) {
        float iv = inv[mi];
        _Float16* dst = xnh + (size_t)(n0 + mi) * DDIM;
        #pragma unroll
        for (int j = 0; j < 8; ++j) {
            int d = lane + 32 * j;
            dst[d] = (_Float16)(tile[d * 33 + mi] * iv);
        }
    }
}

// ---------------------------------------------------------------
// K3: fused sim-GEMM + argmax.  256 threads = 8 waves.
// Each wave: 16 tokens, all 8192 codes. A cached in VGPRs (8 frags),
// cnT staged in LDS as [32 d][128 k] tiles (row stride 136 halves).
// ---------------------------------------------------------------
__global__ void __launch_bounds__(256, 1)
k_simargmax(const _Float16* __restrict__ xnh,
            const _Float16* __restrict__ cnT,
            int* __restrict__ idxb) {
    __shared__ __align__(16) _Float16 lt[32 * 136];
    const int tid  = threadIdx.x;
    const int lane = tid & 31;
    const int wv   = tid >> 5;
    const int n0   = (blockIdx.x * 8 + wv) * 16;
    const int hi   = (lane >> 4) & 1;        // lane half
    const int row  = n0 + (lane & 15);       // token row (A M-index)

    // Preload A fragments: ISA 16-bit A layout -> per lane two 16B chunks
    // at element offsets {0|8} and +16 within each 32-wide d chunk.
    F16x16 A[8];
    #pragma unroll
    for (int c = 0; c < 8; ++c) {
        const v8h* ap = (const v8h*)(xnh + (size_t)row * DDIM + c * 32 + hi * 8);
        A[c].h[0] = ap[0];     // K = c*32 + hi*8 + 0..7
        A[c].h[1] = ap[2];     // K = c*32 + 16 + hi*8 + 0..7
    }

    const v8f vzero = {0.f, 0.f, 0.f, 0.f, 0.f, 0.f, 0.f, 0.f};
    v8f acc[8];
    #pragma unroll
    for (int i = 0; i < 8; ++i) acc[i] = vzero;
    float rmax[8];
    int   ridx[8];
    #pragma unroll
    for (int v = 0; v < 8; ++v) { rmax[v] = -3.4e38f; ridx[v] = 0; }

    const int r  = tid >> 3;          // 0..31  d-row for LDS fill
    const int cc = (tid & 7) * 16;    // 0..112 k-col for LDS fill
    const int col = lane & 15;        // C-matrix column (code offset)

    for (int k0 = 0; k0 < KCODES; k0 += 128) {
        #pragma unroll
        for (int d0 = 0; d0 < DDIM; d0 += 32) {
            __syncthreads();
            const v8h* gp = (const v8h*)(cnT + (size_t)(d0 + r) * KCODES + k0 + cc);
            *(v8h*)&lt[r * 136 + cc]     = gp[0];
            *(v8h*)&lt[r * 136 + cc + 8] = gp[1];
            __syncthreads();
            const int ch = d0 >> 5;
            #pragma unroll
            for (int ks = 0; ks < 8; ++ks) {
                // B layout: lane%16 = B-column (code), elements over d in table order.
                F16x16 B;
                const int bc = ks * 16 + col;
                #pragma unroll
                for (int s = 0; s < 8; ++s) {
                    B.h[0][s] = lt[(hi * 8 + s) * 136 + bc];
                    B.h[1][s] = lt[(16 + hi * 8 + s) * 136 + bc];
                }
                acc[ks] = __builtin_amdgcn_wmma_f32_16x16x32_f16(
                    false, A[ch].v, false, B.v, (short)0, acc[ks], false, false);
            }
        }
        // fold this 128-code strip into the running argmax, reset accum
        #pragma unroll
        for (int ks = 0; ks < 8; ++ks) {
            int code = k0 + ks * 16 + col;
            #pragma unroll
            for (int v = 0; v < 8; ++v) {
                float sv = acc[ks][v];
                if (sv > rmax[v] || (sv == rmax[v] && code < ridx[v])) {
                    rmax[v] = sv; ridx[v] = code;
                }
            }
            acc[ks] = vzero;
        }
    }

    // reduce across the 16 column lanes (xor masks stay inside each half)
    #pragma unroll
    for (int m = 1; m < 16; m <<= 1) {
        #pragma unroll
        for (int v = 0; v < 8; ++v) {
            float om = __shfl_xor(rmax[v], m, 32);
            int   oi = __shfl_xor(ridx[v], m, 32);
            if (om > rmax[v] || (om == rmax[v] && oi < ridx[v])) {
                rmax[v] = om; ridx[v] = oi;
            }
        }
    }
    if ((lane & 15) == 0) {
        int base = n0 + hi * 8;     // C layout: token = v + 8*half
        #pragma unroll
        for (int v = 0; v < 8; ++v) idxb[base + v] = ridx[v];
    }
}

// ---------------------------------------------------------------
// K4: gather q0 = cn[idx], Householder-style rotation, loss.
// One thread per token; lanes sweep l -> coalesced x/q accesses.
// Uses  ||xn+q0||^2 = 2 + 2*(x.q0)/||x||  so only 2 scalars needed.
// ---------------------------------------------------------------
__global__ void k_rotate(const float* __restrict__ x,
                         const float* __restrict__ cn,
                         const int* __restrict__ idxb,
                         float* __restrict__ qout,
                         float* __restrict__ loss) {
    const int n = blockIdx.x * 256 + threadIdx.x;
    const int b = n / LLEN, l = n % LLEN;
    const float* xb = x + (size_t)b * DDIM * LLEN + l;
    const float* q0 = cn + (size_t)idxb[n] * DDIM;
    float s1 = 0.f, s2 = 0.f;
    #pragma unroll 4
    for (int d = 0; d < DDIM; ++d) {
        float xv = xb[(size_t)d * LLEN];
        float qv = q0[d];
        s1 += xv * xv;
        s2 += xv * qv;
    }
    float normx = sqrtf(s1);
    float invn  = 1.0f / fmaxf(normx, 1e-12f);
    float t     = s2 * invn;                                   // xn . q0
    float nw    = 1.0f / fmaxf(sqrtf(fmaxf(2.f + 2.f * t, 0.f)), 1e-12f);
    float dxw   = (normx + s2) * nw;                           // x . w
    float* qo = qout + (size_t)b * DDIM * LLEN + l;
    float lp = 0.f;
    #pragma unroll 4
    for (int d = 0; d < DDIM; ++d) {
        float xv = xb[(size_t)d * LLEN];
        float qv = q0[d];
        float wv = (xv * invn + qv) * nw;
        float rv = (xv - 2.f * dxw * wv + 2.f * normx * qv) * invn;
        qo[(size_t)d * LLEN] = rv;
        float df = rv - xv;
        lp += df * df;
    }
    __shared__ float red[256];
    red[threadIdx.x] = lp;
    __syncthreads();
    for (int s = 128; s > 0; s >>= 1) {
        if (threadIdx.x < s) red[threadIdx.x] += red[threadIdx.x + s];
        __syncthreads();
    }
    if (threadIdx.x == 0)
        atomicAdd(loss, red[0] * (0.25f / (float)(NTOK * DDIM)));
}

// ---------------------------------------------------------------
extern "C" void kernel_launch(void* const* d_in, const int* in_sizes, int n_in,
                              void* d_out, int out_size, void* d_ws, size_t ws_size,
                              hipStream_t stream) {
    const float* x  = (const float*)d_in[0];   // [4,256,4096]
    const float* cb = (const float*)d_in[1];   // [8192,256]
    float* qout = (float*)d_out;               // [4,256,4096] then 1 loss
    float* loss = qout + (size_t)NTOK * DDIM;

    // workspace carve (~21 MB)
    float*    cn   = (float*)d_ws;                               // K*D f32
    _Float16* cnT  = (_Float16*)(cn + (size_t)KCODES * DDIM);    // D*K f16
    _Float16* xnh  = cnT + (size_t)DDIM * KCODES;                // N*D f16
    int*      idxb = (int*)(xnh + (size_t)NTOK * DDIM);          // N  i32

    k_cbnorm   <<<KCODES / 32, 256, 0, stream>>>(cb, cn, cnT, loss);
    k_xnorm    <<<NTOK / 32,   256, 0, stream>>>(x, xnh);
    k_simargmax<<<NTOK / 128,  256, 0, stream>>>(xnh, cnT, idxb);
    k_rotate   <<<NTOK / 256,  256, 0, stream>>>(x, cn, idxb, qout, loss);
}
